// CRF_21148418966035
// MI455X (gfx1250) — matrix operand order repaired and checked by reading, not compiled, not run on hardware
//
#include <hip/hip_runtime.h>
#include <hip/hip_bf16.h>
#include <math.h>

#define B_SZ 256
#define S_SZ 1024
#define T_SZ 128
#define NTAG 126        // start = 126, end = 127
#define NEGV -10000.0f
#define NEG_INF -3.402823466e38f
#define LN2F 0.6931471805599453f
#define PF_DIST 8       // prefetch distance in scan steps

typedef __attribute__((ext_vector_type(16))) _Float16 v16h;
typedef __attribute__((ext_vector_type(8)))  _Float16 v8h;
typedef __attribute__((ext_vector_type(8)))  float    v8f;

union V16H { v16h v; v8h h[2]; };

// ---------------------------------------------------------------------------
// Kernel 1: Et[j][i] = exp(transitions[i][j])  (f16, transposed for B-frags)
// ---------------------------------------------------------------------------
__global__ __launch_bounds__(256) void crf_exp_trans(
    const float* __restrict__ trans, _Float16* __restrict__ Et) {
  int idx = blockIdx.x * 256 + threadIdx.x;        // 64 blocks * 256 = 16384
  int i = idx / T_SZ, j = idx % T_SZ;
  Et[j * T_SZ + i] = (_Float16)__expf(trans[idx]); // exp(-10000) -> 0
}

// ---------------------------------------------------------------------------
// Kernel 2: forward scan. 16 blocks x 256 threads (8 waves).
// Block handles 16 batch rows; wave w owns output columns [16w, 16w+16).
// fv state (16x128 f32) lives in LDS; E B-fragments live in registers.
// Emissions are register-double-buffered 1 step ahead + L2-prefetched
// PF_DIST steps ahead to keep the sequential chain off HBM latency.
// ---------------------------------------------------------------------------
__global__ __launch_bounds__(256) void crf_forward_scan(
    const float* __restrict__ emissions,   // [B, S, T]
    const float* __restrict__ transitions, // [T, T]
    const _Float16* __restrict__ Et,       // [T, T] transposed exp
    float* __restrict__ partition) {       // [B]
  __shared__ __align__(16) float    fv[16][132];
  __shared__ __align__(16) _Float16 af[16][136];   // exp(fv - m) in f16
  __shared__ float mrow[16];

  const int tid  = threadIdx.x;
  const int lane = tid & 31;
  const int wave = tid >> 5;
  const int half = lane >> 4;          // 0: lanes 0-15, 1: lanes 16-31
  const int ln16 = lane & 15;
  const int b0   = blockIdx.x * 16;

  // init fv0: NEG everywhere except start tag (col 126) = 0
  {
    int r  = tid >> 4;
    int c0 = (tid & 15) * 8;
#pragma unroll
    for (int c = 0; c < 8; ++c)
      fv[r][c0 + c] = ((c0 + c) == NTAG) ? 0.0f : NEGV;
  }

  // Preload loop-invariant B fragments of E for this wave's 16 columns.
  // B 32x16 frag in this code's contiguous k order (same as A): lane holds N = ln16; K = half*16 + (0..15) per k-step.
  V16H bfrag[4];
  {
    const _Float16* erow = Et + (size_t)(wave * 16 + ln16) * T_SZ;
#pragma unroll
    for (int k = 0; k < 4; ++k) {
      int K0 = k * 32 + half * 16;
      bfrag[k].h[0] = *(const v8h*)(erow + K0);
      bfrag[k].h[1] = *(const v8h*)(erow + K0 + 8);
    }
  }

  const float* emBase = emissions + (size_t)b0 * S_SZ * T_SZ;
  const int ccol = wave * 16 + ln16;   // this lane's output column (C-layout N)

  // emissions double buffer: emv holds step s, emn is loaded for step s+1
  float emv[8], emn[8];
#pragma unroll
  for (int v = 0; v < 8; ++v) {
    int r = v + half * 8;
    emv[v] = emBase[(size_t)r * S_SZ * T_SZ + (size_t)1 * T_SZ + ccol];
  }

  for (int s = 1; s < S_SZ; ++s) {
    __syncthreads();
    // ---- phase 1: per-row max (shfl butterfly) + exp(fv - m) -> f16 LDS ----
    {
      int r  = tid >> 4;               // 16 threads per row, contiguous 16-lane half
      int c0 = (tid & 15) * 8;
      float vals[8];
      float mx = NEG_INF;
#pragma unroll
      for (int c = 0; c < 8; ++c) { vals[c] = fv[r][c0 + c]; mx = fmaxf(mx, vals[c]); }
#pragma unroll
      for (int m = 8; m >= 1; m >>= 1) mx = fmaxf(mx, __shfl_xor(mx, m, 32));
      if ((tid & 15) == 0) mrow[r] = mx;
      _Float16 a8[8];
#pragma unroll
      for (int c = 0; c < 8; ++c) a8[c] = (_Float16)__expf(vals[c] - mx);
      *(v8h*)(&af[r][c0]) = *(const v8h*)a8;
    }

    // ---- pipelined emissions: register load for s+1, L2 prefetch for s+PF ----
    {
      int sn = (s + 1 < S_SZ) ? s + 1 : s;
#pragma unroll
      for (int v = 0; v < 8; ++v) {
        int r = v + half * 8;
        emn[v] = emBase[(size_t)r * S_SZ * T_SZ + (size_t)sn * T_SZ + ccol];
      }
      if (tid < 64 && s + PF_DIST < S_SZ) {
        int r = tid >> 2, ln = tid & 3;   // 16 rows x 4 cachelines (512B/row)
        __builtin_prefetch(
            emBase + (size_t)r * S_SZ * T_SZ + (size_t)(s + PF_DIST) * T_SZ + ln * 32,
            0, 0);
      }
    }
    __syncthreads();

    // ---- phase 2: [16x128]@[128x128] tile via 4x v_wmma_f32_16x16x32_f16 ----
    // Issue all 8 A-fragment ds_load_b128 first, then chain the WMMAs.
    V16H a[4];
#pragma unroll
    for (int k = 0; k < 4; ++k) {
      // A 16x32 frag in this code's contiguous k order (same as B): lane holds M = ln16;
      // K = k*32 + half*16 + (0..15)
      const _Float16* arow = &af[ln16][0];
      int K0 = k * 32 + half * 16;
      a[k].h[0] = *(const v8h*)(arow + K0);
      a[k].h[1] = *(const v8h*)(arow + K0 + 8);
    }
    v8f acc = {};
#pragma unroll
    for (int k = 0; k < 4; ++k)
      acc = __builtin_amdgcn_wmma_f32_16x16x32_f16(
          false, a[k].v, false, bfrag[k].v, (short)0, acc, false, false);

    // ---- phase 3: fv_new = ln2*log2(acc) + m + emit[s] ----
    // acc is a sum of non-negative products; dominant term exp(0)*E >= tiny
    // normal, or exactly 0 (-> -inf), so raw v_log_f32 is safe (no denorm guard).
    // C layout: VGPR v -> row (v + half*8), col ccol.
#pragma unroll
    for (int v = 0; v < 8; ++v) {
      int r = v + half * 8;
      float me = mrow[r] + emv[v];
      fv[r][ccol] = __builtin_fmaf(__builtin_amdgcn_logf(acc[v]), LN2F, me);
    }
#pragma unroll
    for (int v = 0; v < 8; ++v) emv[v] = emn[v];
  }

  __syncthreads();
  // ---- finalize: partition[b] = logsumexp_j(fv[b,j] + transitions[end, j]) ----
  {
    int r  = tid >> 4;
    int c0 = (tid & 15) * 8;
    const float* tend = transitions + (NTAG + 1) * T_SZ;
    float vals[8];
    float mx = NEG_INF;
#pragma unroll
    for (int c = 0; c < 8; ++c) {
      vals[c] = fv[r][c0 + c] + tend[c0 + c];
      mx = fmaxf(mx, vals[c]);
    }
#pragma unroll
    for (int m = 8; m >= 1; m >>= 1) mx = fmaxf(mx, __shfl_xor(mx, m, 32));
    float sum = 0.0f;
#pragma unroll
    for (int c = 0; c < 8; ++c) sum += __expf(vals[c] - mx);
#pragma unroll
    for (int m = 8; m >= 1; m >>= 1) sum += __shfl_xor(sum, m, 32);
    if ((tid & 15) == 0) partition[b0 + r] = __logf(sum) + mx;
  }
}

// ---------------------------------------------------------------------------
// Kernel 3: gold path score. One block per batch element.
// ---------------------------------------------------------------------------
__global__ __launch_bounds__(256) void crf_gold(
    const float* __restrict__ emissions, const int* __restrict__ tags,
    const float* __restrict__ transitions, float* __restrict__ gold) {
  __shared__ float lt[T_SZ * T_SZ];    // 64 KB transition table
  __shared__ float red[8];
  const int b = blockIdx.x, tid = threadIdx.x;
  for (int i = tid; i < T_SZ * T_SZ; i += 256) lt[i] = transitions[i];
  __syncthreads();
  const int*   tg = tags + (size_t)b * S_SZ;
  const float* em = emissions + (size_t)b * S_SZ * T_SZ;
  float acc = 0.0f;
#pragma unroll
  for (int k = 0; k < 4; ++k) {
    int s  = tid + k * 256;
    int t0 = tg[s];
    acc += em[(size_t)s * T_SZ + t0];
    if (s < S_SZ - 1) acc += lt[t0 * T_SZ + tg[s + 1]];
  }
  if (tid == 0)
    acc += lt[NTAG * T_SZ + tg[0]] + lt[tg[S_SZ - 1] * T_SZ + (NTAG + 1)];
#pragma unroll
  for (int m = 16; m >= 1; m >>= 1) acc += __shfl_xor(acc, m, 32);
  if ((tid & 31) == 0) red[tid >> 5] = acc;
  __syncthreads();
  if (tid == 0) {
    float t = 0.0f;
#pragma unroll
    for (int w = 0; w < 8; ++w) t += red[w];
    gold[b] = t;
  }
}

// ---------------------------------------------------------------------------
// Kernel 4: loss = mean(partition - gold)
// ---------------------------------------------------------------------------
__global__ __launch_bounds__(256) void crf_finalize(
    const float* __restrict__ partition, const float* __restrict__ gold,
    float* __restrict__ out) {
  __shared__ float red[8];
  const int tid = threadIdx.x;
  float d = partition[tid] - gold[tid];
#pragma unroll
  for (int m = 16; m >= 1; m >>= 1) d += __shfl_xor(d, m, 32);
  if ((tid & 31) == 0) red[tid >> 5] = d;
  __syncthreads();
  if (tid == 0) {
    float t = 0.0f;
#pragma unroll
    for (int w = 0; w < 8; ++w) t += red[w];
    out[0] = t / (float)B_SZ;
  }
}

// ---------------------------------------------------------------------------
extern "C" void kernel_launch(void* const* d_in, const int* in_sizes, int n_in,
                              void* d_out, int out_size, void* d_ws, size_t ws_size,
                              hipStream_t stream) {
  (void)in_sizes; (void)n_in; (void)out_size; (void)ws_size;
  const float* emissions   = (const float*)d_in[0];
  const int*   tags        = (const int*)d_in[1];
  const float* transitions = (const float*)d_in[2];
  float* out = (float*)d_out;

  char* ws = (char*)d_ws;
  _Float16* Et        = (_Float16*)ws;           // 128*128*2 = 32768 B
  float*    partition = (float*)(ws + 32768);    // 1024 B
  float*    gold      = (float*)(ws + 32768 + 1024);

  crf_exp_trans   <<<64, 256, 0, stream>>>(transitions, Et);
  crf_forward_scan<<<16, 256, 0, stream>>>(emissions, transitions, Et, partition);
  crf_gold        <<<B_SZ, 256, 0, stream>>>(emissions, tags, transitions, gold);
  crf_finalize    <<<1, 256, 0, stream>>>(partition, gold, out);
}
